// ESP_Fi_LSTM_33303176413513
// MI455X (gfx1250) — matrix operand order repaired and checked by reading, not compiled, not run on hardware
//
#include <hip/hip_runtime.h>

typedef __bf16 bf16_t;
typedef bf16_t v16bf __attribute__((ext_vector_type(16)));
typedef float  v8f   __attribute__((ext_vector_type(8)));

#define HDIM    128
#define KCAT    192     // 64 (F=52 padded) + 128 (H)
#define TSTEPS  512
#define BTILE   16
#define THREADS 512
#define XROW_STRIDE 106496u  // 2048*52
#define XT_STRIDE   208u     // 4*52

__device__ __forceinline__ unsigned short f2bf(float f) {
  unsigned u = __float_as_uint(f);
  u += 0x7fffu + ((u >> 16) & 1u);   // round-to-nearest-even
  return (unsigned short)(u >> 16);
}

__device__ __forceinline__ float fast_sig(float v) {
  return __builtin_amdgcn_rcpf(1.0f + __expf(-v));
}

__device__ __forceinline__ float fast_tanh(float v) {
#if __has_builtin(__builtin_amdgcn_tanhf)
  return __builtin_amdgcn_tanhf(v);          // native V_TANH_F32 on gfx1250
#else
  const float e = __expf(2.0f * v);          // tanh(x) = 1 - 2/(e^2x + 1)
  return 1.0f - 2.0f * __builtin_amdgcn_rcpf(e + 1.0f);
#endif
}

union BF { v16bf v; unsigned short s[16]; };
union AF { v16bf v; uint4 q[2]; };

__global__ __launch_bounds__(THREADS) void lstm_fused_kernel(
    const float* __restrict__ x,   const float* __restrict__ Wih,
    const float* __restrict__ Whh, const float* __restrict__ bih,
    const float* __restrict__ bhh, const float* __restrict__ Wfc,
    const float* __restrict__ bfc, float* __restrict__ out)
{
  // ping-pong [x_t | h_t] bf16 A panels (kills WAR hazard -> 1 barrier/step)
  __shared__ __align__(16) unsigned short panel[2][BTILE * KCAT];
  // wave-private raw-gate buffers: [wave][m=16][lc=32] f32, lc = unit*4 + {i,f,g,o}
  __shared__ __align__(16) float gbuf[16][BTILE * 32];
  __shared__ float hsL[BTILE * HDIM];   // final h-sum for the FC epilogue

  const int tid  = threadIdx.x;
  const int lane = tid & 31;
  const int wid  = tid >> 5;           // 16 waves; wave w owns hidden units w*8..w*8+7
  const int b0   = blockIdx.x * BTILE;

  const int ncol  = lane & 15;
  const int khalf = (lane >> 4) << 4;

  // ---- B fragments: permuted gate columns, bf16, pinned in 96 VGPRs ----
  // permuted column lc = sub*16 + ncol -> unit u = lc>>2, gate gt = lc&3,
  // original gate row g = gt*128 + (wid*8 + u)
  BF bfr[2][6];
  float bias[2];
  #pragma unroll
  for (int sub = 0; sub < 2; ++sub) {
    const int lc = sub * 16 + ncol;
    const int g  = (lc & 3) * 128 + wid * 8 + (lc >> 2);
    bias[sub] = bih[g] + bhh[g];
    #pragma unroll
    for (int kt = 0; kt < 6; ++kt) {
      #pragma unroll
      for (int i = 0; i < 16; ++i) {
        const int k = kt * 32 + khalf + i;
        float w;
        if (k < 52)      w = Wih[g * 52 + k];
        else if (k < 64) w = 0.f;
        else             w = Whh[g * 128 + (k - 64)];
        bfr[sub][kt].s[i] = f2bf(w);
      }
    }
  }

  // ---- init panels: zero (h0 = 0, K padding), then stage x(t=0) ----
  for (int i = tid; i < 2 * BTILE * KCAT; i += THREADS)
    ((unsigned short*)panel)[i] = 0;
  __syncthreads();
  {
    const size_t rb = (size_t)(b0 + wid) * XROW_STRIDE;  // wave w stages batch row w
    for (int i = lane; i < 52; i += 32)
      panel[0][wid * KCAT + i] = f2bf(x[rb + i]);
  }
  __syncthreads();

  // A layout: lane<16: row=lane, K={0..15}+32*kt ; lane>=16: row=lane-16, +16
  const int arow  = ncol;
  const int akoff = (lane >> 4) * 16;
  const int mrow0 = (lane >> 4) * 8;

  float creg[4] = {0.f, 0.f, 0.f, 0.f};   // cell state, 4 (m,u) pairs per lane
  float hreg[4] = {0.f, 0.f, 0.f, 0.f};   // running h sum
  int p = 0;

  for (int t = 0; t < TSTEPS; ++t) {
    // ---- gates = [x_t | h] @ Wcat^T + b (bias folded into C init) ----
    v8f c0, c1;
    #pragma unroll
    for (int j = 0; j < 8; ++j) { c0[j] = bias[0]; c1[j] = bias[1]; }
    const unsigned short* ppan = panel[p];
    #pragma unroll
    for (int kt = 0; kt < 6; ++kt) {
      AF a;
      const unsigned short* ap = &ppan[arow * KCAT + kt * 32 + akoff];
      a.q[0] = *reinterpret_cast<const uint4*>(ap);
      a.q[1] = *reinterpret_cast<const uint4*>(ap + 8);
      c0 = __builtin_amdgcn_wmma_f32_16x16x32_bf16(false, a.v, false, bfr[0][kt].v,
                                                   (short)0, c0, false, false);
      c1 = __builtin_amdgcn_wmma_f32_16x16x32_bf16(false, a.v, false, bfr[1][kt].v,
                                                   (short)0, c1, false, false);
    }

    // ---- wave-private gate scatter (no block barrier needed) ----
    float* gb = gbuf[wid];
    #pragma unroll
    for (int j = 0; j < 8; ++j) {
      gb[(mrow0 + j) * 32 + ncol]      = c0[j];
      gb[(mrow0 + j) * 32 + 16 + ncol] = c1[j];
    }
    __builtin_amdgcn_wave_barrier();   // compiler fence; DS is in-order per wave

    // ---- cell update: 4 (m,u) pairs per lane; {i,f,g,o} is one b128 load ----
    unsigned short* hout = panel[p ^ 1];
    #pragma unroll
    for (int rep = 0; rep < 4; ++rep) {
      const int idx = rep * 32 + lane;
      const int m = idx & 15;
      const int u = idx >> 4;
      const float4 gv = *reinterpret_cast<const float4*>(&gb[m * 32 + u * 4]);
      const float si = fast_sig(gv.x);
      const float sf = fast_sig(gv.y);
      const float tg = fast_tanh(gv.z);
      const float so = fast_sig(gv.w);
      const float cc = sf * creg[rep] + si * tg;
      const float hh = so * fast_tanh(cc);
      creg[rep] = cc;
      hreg[rep] += hh;
      hout[m * KCAT + 64 + wid * 8 + u] = f2bf(hh);   // h feeds next step's panel
    }

    // ---- stage x(t+1) into the other panel; prefetch t+2 ----
    if (t + 1 < TSTEPS) {
      const size_t rb = (size_t)(b0 + wid) * XROW_STRIDE + (size_t)(t + 1) * XT_STRIDE;
      for (int i = lane; i < 52; i += 32) {
        hout[wid * KCAT + i] = f2bf(x[rb + i]);
        if (t + 2 < TSTEPS) __builtin_prefetch(&x[rb + XT_STRIDE + i], 0, 0);
      }
    }
    __syncthreads();   // single barrier per step
    p ^= 1;
  }

  // ---- epilogue: h-sum -> LDS, then FC: out = mean(h) @ W_fc^T + b_fc ----
  #pragma unroll
  for (int rep = 0; rep < 4; ++rep) {
    const int idx = rep * 32 + lane;
    hsL[(idx & 15) * HDIM + wid * 8 + (idx >> 4)] = hreg[rep];
  }
  __syncthreads();
  if (tid < BTILE * 7) {
    const int m  = tid / 7;
    const int cc = tid - m * 7;
    float acc = 0.f;
    const float* wr = Wfc + cc * HDIM;
    const float* hr = hsL + m * HDIM;
    #pragma unroll 8
    for (int k = 0; k < HDIM; ++k) acc += hr[k] * wr[k];
    out[(b0 + m) * 7 + cc] = acc * (1.0f / (float)TSTEPS) + bfc[cc];
  }
}

extern "C" void kernel_launch(void* const* d_in, const int* in_sizes, int n_in,
                              void* d_out, int out_size, void* d_ws, size_t ws_size,
                              hipStream_t stream) {
  const float* x   = (const float*)d_in[0];
  const float* Wih = (const float*)d_in[1];
  const float* Whh = (const float*)d_in[2];
  const float* bih = (const float*)d_in[3];
  const float* bhh = (const float*)d_in[4];
  const float* Wfc = (const float*)d_in[5];
  const float* bfc = (const float*)d_in[6];
  float* out = (float*)d_out;

  lstm_fused_kernel<<<dim3(256 / BTILE), dim3(THREADS), 0, stream>>>(
      x, Wih, Whh, bih, bhh, Wfc, bfc, out);
}
